// VectorQuantize_67044439490971
// MI455X (gfx1250) — matrix operand order repaired and checked by reading, not compiled, not run on hardware
//
#include <hip/hip_runtime.h>

typedef __attribute__((ext_vector_type(16))) __bf16        v16bf;
typedef __attribute__((ext_vector_type(8)))  float         v8f;
typedef __attribute__((ext_vector_type(4)))  unsigned int  u32x4;
typedef __attribute__((ext_vector_type(4)))  int           i32x4;
typedef __attribute__((ext_vector_type(8)))  int           i32x8;
typedef __attribute__((ext_vector_type(4)))  float         f32x4;

#define N_TOK  32768   // B*S = 16*2048
#define DIM    512
#define CODES  8192
#define SEQ    2048
#define BM     128     // tokens per workgroup
#define BN     128     // codes per N-iteration
#define BK     32      // K chunk (bf16 WMMA K)
#define NCH    (DIM / BK)       // 16 K-chunks
#define NITER  (CODES / BN)     // 64 N-iterations
#define TOTCH  (NITER * NCH)    // 1024 chunks total

// ---------- bf16 split helpers (round-to-nearest-even) ----------
__device__ __forceinline__ unsigned int f32_to_bf16_rn(float f) {
  unsigned int u = __float_as_uint(f);
  unsigned int lsb = (u >> 16) & 1u;
  u += 0x7fffu + lsb;
  return u >> 16;
}
__device__ __forceinline__ float bf16_to_f32(unsigned int h) {
  return __uint_as_float(h << 16);
}
__device__ __forceinline__ void split4(f32x4 v, unsigned int hp[2], unsigned int lp[2]) {
  float f0 = v.x, f1 = v.y, f2 = v.z, f3 = v.w;
  unsigned int h0 = f32_to_bf16_rn(f0), h1 = f32_to_bf16_rn(f1);
  unsigned int h2 = f32_to_bf16_rn(f2), h3 = f32_to_bf16_rn(f3);
  unsigned int l0 = f32_to_bf16_rn(f0 - bf16_to_f32(h0));
  unsigned int l1 = f32_to_bf16_rn(f1 - bf16_to_f32(h1));
  unsigned int l2 = f32_to_bf16_rn(f2 - bf16_to_f32(h2));
  unsigned int l3 = f32_to_bf16_rn(f3 - bf16_to_f32(h3));
  hp[0] = h0 | (h1 << 16); hp[1] = h2 | (h3 << 16);
  lp[0] = l0 | (l1 << 16); lp[1] = l2 | (l3 << 16);
}

union FragU { v16bf v; u32x4 q[2]; };

// ---------- TDM: 2D tile load (tile BKxBN of bf16, tensor row len DIM) ----------
__device__ __forceinline__ void tdm_load_tile(unsigned lds_off, const void* gaddr) {
  unsigned long long ga = (unsigned long long)gaddr;
  u32x4 g0;
  g0.x = 1u;                                   // count=1, user mode
  g0.y = lds_off;                              // lds_addr (bytes)
  g0.z = (unsigned)ga;                         // global_addr[31:0]
  g0.w = (unsigned)(ga >> 32) | (2u << 30);    // global_addr[56:32] | type=2
  i32x8 g1;
  g1[0] = 0x10000;                 // workgroup_mask=0, data_size=1 (2 bytes)
  g1[1] = (int)(512u << 16);       // tensor_dim0 = 512 (low16 in [31:16])
  g1[2] = (int)(8192u << 16);      // tensor_dim0 hi=0 | tensor_dim1=8192 low16
  g1[3] = (int)(32u << 16);        // tensor_dim1 hi=0 | tile_dim0 = 32
  g1[4] = 128;                     // tile_dim1 = 128, tile_dim2 = 0
  g1[5] = 512;                     // tensor_dim0_stride = 512 (low32)
  g1[6] = 0;                       // stride hi, tensor_dim1_stride low16
  g1[7] = 0;
  i32x4 z4 = {0, 0, 0, 0};
#if __clang_major__ >= 23
  i32x8 z8 = {0, 0, 0, 0, 0, 0, 0, 0};
  __builtin_amdgcn_tensor_load_to_lds(g0, g1, z4, z4, z8, 0);
#else
  __builtin_amdgcn_tensor_load_to_lds(g0, g1, z4, z4, 0);
#endif
}

// ---------- kernel 0: e2[c] = sum_d embed[c,d]^2 ----------
__global__ __launch_bounds__(256) void vq_e2_kernel(const float* __restrict__ embed,
                                                    float* __restrict__ e2g) {
  int c = blockIdx.x * 256 + threadIdx.x;
  const f32x4* p = (const f32x4*)(embed + (size_t)c * DIM);
  float s = 0.0f;
  #pragma unroll 8
  for (int i = 0; i < DIM / 4; ++i) {
    f32x4 v = p[i];
    s += v.x * v.x + v.y * v.y + v.z * v.z + v.w * v.w;
  }
  e2g[c] = s;
}

// ---------- kernel 1: pre-split embed into bf16 hi/lo planes ----------
__global__ __launch_bounds__(256) void vq_split_kernel(const float* __restrict__ embed,
                                                       unsigned short* __restrict__ Eh,
                                                       unsigned short* __restrict__ El) {
  int f4 = blockIdx.x * 256 + threadIdx.x;       // over CODES*DIM/4 float4s
  f32x4 v = *(const f32x4*)(embed + (size_t)f4 * 4);
  unsigned int hp[2], lp[2];
  split4(v, hp, lp);
  ((unsigned int*)Eh)[f4 * 2 + 0] = hp[0];
  ((unsigned int*)Eh)[f4 * 2 + 1] = hp[1];
  ((unsigned int*)El)[f4 * 2 + 0] = lp[0];
  ((unsigned int*)El)[f4 * 2 + 1] = lp[1];
}

// ---------- kernel 2: streaming argmin via bf16x3 WMMA + TDM double buffer ----------
__global__ __launch_bounds__(256) void vq_argmin_kernel(const float* __restrict__ x,
                                                        const unsigned short* __restrict__ EhW,
                                                        const unsigned short* __restrict__ ElW,
                                                        const float* __restrict__ e2g,
                                                        int* __restrict__ bestIdx) {
  __shared__ union {
    struct {
      alignas(16) unsigned short Xh[BM][DIM];       // 128 KB
      alignas(16) unsigned short Xl[BM][DIM];       // 128 KB
      alignas(16) unsigned short E[2][2][BN][BK];   // 32 KB [buf][hi/lo][code][k]
    } t;
    struct {
      float d[BM][16];                               // 8 KB
      int   i[BM][16];                               // 8 KB
    } r;
  } lds;

  const int tid     = threadIdx.x;
  const int lane    = tid & 31;
  const int wave    = tid >> 5;          // 8 waves = 8 row strips of 16 tokens
  const int halfSel = lane >> 4;
  const int l15     = lane & 15;
  const int tokBase = blockIdx.x * BM;

  // wave 0 primes the TDM pipeline (chunks 0 and 1) before the X prologue,
  // so the DMA overlaps the conversion work below.
  if (wave == 0) {
    #pragma unroll
    for (int c0 = 0; c0 < 2; ++c0) {
      int nb = (c0 >> 4) * BN, kc = (c0 & (NCH - 1)) * BK, buf = c0 & 1;
      tdm_load_tile((unsigned)(unsigned long long)&lds.t.E[buf][0][0][0],
                    EhW + (size_t)nb * DIM + kc);
      tdm_load_tile((unsigned)(unsigned long long)&lds.t.E[buf][1][0][0],
                    ElW + (size_t)nb * DIM + kc);
    }
  }

  // ---- X prologue: convert this WG's 128 tokens to split bf16 once ----
  #pragma unroll 4
  for (int r = 0; r < (BM * DIM / 4) / 256; ++r) {   // 64 float4 per thread
    int f4 = tid + 256 * r;
    int m  = f4 >> 7;            // 128 float4 per row
    int kq = f4 & 127;
    f32x4 v = *(const f32x4*)(x + (size_t)(tokBase + m) * DIM + kq * 4);
    unsigned int hp[2], lp[2];
    split4(v, hp, lp);
    ((unsigned int*)&lds.t.Xh[m][kq * 4])[0] = hp[0];
    ((unsigned int*)&lds.t.Xh[m][kq * 4])[1] = hp[1];
    ((unsigned int*)&lds.t.Xl[m][kq * 4])[0] = lp[0];
    ((unsigned int*)&lds.t.Xl[m][kq * 4])[1] = lp[1];
  }
  __syncthreads();

  float bestd[8];
  int   besti[8];
  #pragma unroll
  for (int j = 0; j < 8; ++j) { bestd[j] = 3.0e38f; besti[j] = 0; }

  int ch = 0;
  for (int nb = 0; nb < CODES; nb += BN) {
    v8f acc[8];
    #pragma unroll
    for (int t = 0; t < 8; ++t)
      #pragma unroll
      for (int j = 0; j < 8; ++j) acc[t][j] = 0.0f;

    for (int kci = 0; kci < NCH; ++kci, ++ch) {
      // ensure this chunk's 2 TDM ops completed (in-order TENSORcnt)
      if (wave == 0) {
        if (ch + 1 < TOTCH) __builtin_amdgcn_s_wait_tensorcnt(2);
        else                __builtin_amdgcn_s_wait_tensorcnt(0);
      }
      __syncthreads();
      const int buf = ch & 1;
      const int kc  = kci * BK;

      // A fragments (16 rows x 32 K), hi and lo
      const int arow = wave * 16 + l15;
      FragU ah, al;
      {
        const u32x4* ph = (const u32x4*)&lds.t.Xh[arow][kc];
        const u32x4* pl = (const u32x4*)&lds.t.Xl[arow][kc];
        ah.q[0] = ph[halfSel]; ah.q[1] = ph[halfSel + 2];
        al.q[0] = pl[halfSel]; al.q[1] = pl[halfSel + 2];
      }

      #pragma unroll
      for (int nt = 0; nt < 8; ++nt) {
        const int brow = nt * 16 + l15;
        FragU bh, bl;
        const u32x4* ph = (const u32x4*)&lds.t.E[buf][0][brow][0];
        const u32x4* pl = (const u32x4*)&lds.t.E[buf][1][brow][0];
        bh.q[0] = ph[halfSel]; bh.q[1] = ph[halfSel + 2];
        bl.q[0] = pl[halfSel]; bl.q[1] = pl[halfSel + 2];

        acc[nt] = __builtin_amdgcn_wmma_f32_16x16x32_bf16(
            false, ah.v, false, bh.v, (short)0, acc[nt], false, false);
        acc[nt] = __builtin_amdgcn_wmma_f32_16x16x32_bf16(
            false, ah.v, false, bl.v, (short)0, acc[nt], false, false);
        acc[nt] = __builtin_amdgcn_wmma_f32_16x16x32_bf16(
            false, al.v, false, bh.v, (short)0, acc[nt], false, false);
      }
      __syncthreads();   // all waves done reading E[buf]

      // refill E[buf] with chunk ch+2
      if (wave == 0 && ch + 2 < TOTCH) {
        int c2 = ch + 2;
        int nb2 = (c2 >> 4) * BN, kc2 = (c2 & (NCH - 1)) * BK, b2 = c2 & 1;
        tdm_load_tile((unsigned)(unsigned long long)&lds.t.E[b2][0][0][0],
                      EhW + (size_t)nb2 * DIM + kc2);
        tdm_load_tile((unsigned)(unsigned long long)&lds.t.E[b2][1][0][0],
                      ElW + (size_t)nb2 * DIM + kc2);
      }
    }

    // score = e2[c] - 2*xy ; strict-< running argmin over ascending code index
    #pragma unroll
    for (int nt = 0; nt < 8; ++nt) {
      int col = nb + nt * 16 + l15;
      float e2v = e2g[col];
      #pragma unroll
      for (int j = 0; j < 8; ++j) {
        float s = e2v - 2.0f * acc[nt][j];
        if (s < bestd[j]) { bestd[j] = s; besti[j] = col; }
      }
    }
  }

  // ---- final per-row reduction: 16 lane-candidates; tie -> smaller index ----
  __syncthreads();
  #pragma unroll
  for (int j = 0; j < 8; ++j) {
    int row = wave * 16 + j + 8 * halfSel;   // C-layout: VGPR j -> M=j / M=j+8
    lds.r.d[row][l15] = bestd[j];
    lds.r.i[row][l15] = besti[j];
  }
  __syncthreads();
  if (tid < BM) {
    float bd = lds.r.d[tid][0];
    int   bi = lds.r.i[tid][0];
    #pragma unroll
    for (int s = 1; s < 16; ++s) {
      float d  = lds.r.d[tid][s];
      int   i2 = lds.r.i[tid][s];
      if (d < bd || (d == bd && i2 < bi)) { bd = d; bi = i2; }
    }
    bestIdx[tokBase + tid] = bi;
  }
}

// ---------- kernel 3: gather + mask writeback ----------
__global__ __launch_bounds__(128) void vq_writeback_kernel(const float* __restrict__ embed,
                                                           const int* __restrict__ bestIdx,
                                                           const int* __restrict__ input_length,
                                                           float* __restrict__ quant,
                                                           int* __restrict__ indOut) {
  int t = blockIdx.x;
  int b = t >> 11;                  // / SEQ
  int s = t & (SEQ - 1);
  bool valid = s < input_length[b];
  int idx = bestIdx[t];
  const f32x4* src = (const f32x4*)(embed + (size_t)idx * DIM);
  f32x4 val;
  if (valid) {
    val = src[threadIdx.x];
  } else {
    val.x = 0.0f; val.y = 0.0f; val.z = 0.0f; val.w = 0.0f;
  }
  ((f32x4*)(quant + (size_t)t * DIM))[threadIdx.x] = val;
  if (threadIdx.x == 0) indOut[t] = valid ? idx : -1;
}

extern "C" void kernel_launch(void* const* d_in, const int* in_sizes, int n_in,
                              void* d_out, int out_size, void* d_ws, size_t ws_size,
                              hipStream_t stream) {
  const float* x            = (const float*)d_in[0];   // [16,2048,512] f32
  const int*   input_length = (const int*)d_in[1];     // [16] i32
  const float* embed        = (const float*)d_in[2];   // [8192,512] f32

  float* quant  = (float*)d_out;
  int*   indOut = (int*)((float*)d_out + (size_t)N_TOK * DIM);

  // workspace: e2 (32 KB) | bestIdx (128 KB) | pad | Eh (8 MB) | El (8 MB)
  float*          e2g     = (float*)d_ws;
  int*            bestIdx = (int*)((char*)d_ws + CODES * 4);
  unsigned short* EhW     = (unsigned short*)((char*)d_ws + (1u << 20));
  unsigned short* ElW     = (unsigned short*)((char*)d_ws + (1u << 20) + (size_t)CODES * DIM * 2);

  vq_e2_kernel<<<CODES / 256, 256, 0, stream>>>(embed, e2g);
  vq_split_kernel<<<(CODES * DIM / 4) / 256, 256, 0, stream>>>(embed, EhW, ElW);
  vq_argmin_kernel<<<N_TOK / BM, 256, 0, stream>>>(x, EhW, ElW, e2g, bestIdx);
  vq_writeback_kernel<<<N_TOK, 128, 0, stream>>>(embed, bestIdx, input_length,
                                                 quant, indOut);
}
